// MagneticRealAgnosticDensityInteractionBlock_42125039239749
// MI455X (gfx1250) — matrix operand order repaired and checked by reading, not compiled, not run on hardware
//
#include <hip/hip_runtime.h>
#include <math.h>

// ---------------------------------------------------------------------------
// MACE-style magnetic density interaction block for gfx1250 (MI455X).
// All GEMM-shaped work runs on v_wmma_f32_16x16x32_bf16 (f32 accumulate).
// GEMM: 4 waves/block, each wave computes a 16x64 strip (4 accumulators,
// one shared A operand per K-chunk -> 4x less A traffic, 4 independent WMMAs).
// Skip-TP: WMMA Weff generation + register-accumulated apply (no LDS RMW).
// ---------------------------------------------------------------------------

typedef __attribute__((ext_vector_type(16))) __bf16 v16bf;
typedef __attribute__((ext_vector_type(8)))  float  v8f;

__device__ __forceinline__ float siluf(float v) { return v / (1.f + __expf(-v)); }

// --------------------------- utility kernels --------------------------------

__global__ void zero_k(float* __restrict__ p, long n) {
  long i = (long)blockIdx.x * blockDim.x + threadIdx.x;
  if (i < n) p[i] = 0.f;
}

// h0[e] = concat(edge_feats[e] (8), magmom_inv[sender[e]] (8), zeros (16))
__global__ void build_h0_k(const float* __restrict__ ef,
                           const float* __restrict__ minv,
                           const int* __restrict__ eidx,
                           float* __restrict__ h0, int E) {
  int i = blockIdx.x * blockDim.x + threadIdx.x;
  if (i >= E * 32) return;
  int e = i >> 5, j = i & 31;
  float v = 0.f;
  if (j < 8)       v = ef[e * 8 + j];
  else if (j < 16) v = minv[(long)eidx[e] * 8 + (j - 8)];
  h0[i] = v;
}

// zero-pad rows: out[rows_out x cols], rows >= rows_in become 0
__global__ void pad_rows_k(const float* __restrict__ in, float* __restrict__ out,
                           int rows_in, int rows_out, int cols) {
  int i = blockIdx.x * blockDim.x + threadIdx.x;
  if (i >= rows_out * cols) return;
  int r = i / cols, c = i % cols;
  out[i] = (r < rows_in) ? in[r * cols + c] : 0.f;
}

// zero-pad cols: [rows x cols_in] -> [rows x cols_out]
__global__ void pad_cols_k(const float* __restrict__ in, float* __restrict__ out,
                           int rows, int cols_in, int cols_out) {
  int i = blockIdx.x * blockDim.x + threadIdx.x;
  if (i >= rows * cols_out) return;
  int r = i / cols_out, c = i % cols_out;
  out[i] = (c < cols_in) ? in[r * cols_in + c] : 0.f;
}

// edge_density -> segment sum over receiver
__global__ void density_k(const float* __restrict__ ef,
                          const float* __restrict__ Wd,
                          const int* __restrict__ eidx,
                          float* __restrict__ density, int E) {
  int e = blockIdx.x * blockDim.x + threadIdx.x;
  if (e >= E) return;
  float t = 0.f;
#pragma unroll
  for (int j = 0; j < 8; ++j) t += ef[e * 8 + j] * Wd[j];
  t *= 0.3535533906f;                    // 1/sqrt(R)
  atomicAdd(&density[eidx[E + e]], tanhf(t * t));
}

// ------------------- generic strided bf16 WMMA GEMM -------------------------
// C[m,n] = act( scale * sum_k A[m,k] * B[k,n] ),  K % 32 == 0 (pre-padded).
// 4 waves/block; each wave owns a 16-row strip and NT column tiles (16 wide).
// One A operand per K-chunk shared by NT WMMAs. ACONTIG -> float4 A loads.
template <bool ACONTIG, int NT>
__global__ void gemm_bf16_wmma(const float* __restrict__ A, int a_rs, int a_cs,
                               const float* __restrict__ B, int b_rs, int b_cs,
                               float* __restrict__ C, int c_rs, int c_cs,
                               int K, float scale, int act) {
  int tid  = threadIdx.x;          // 0..127
  int wave = tid >> 5;
  int lane = tid & 31;
  bool hi  = lane >= 16;
  int lid  = lane & 15;
  int row0 = (blockIdx.x * 4 + wave) * 16;
  int col0 = blockIdx.y * (16 * NT);
  int asel = hi ? 8 : 0;    // A K-offset for this half-wave
  int bsel = hi ? 8 : 0;    // B K-offset

  const float* Ar = A + (size_t)(row0 + lid) * a_rs;

  v8f acc[NT];
#pragma unroll
  for (int t = 0; t < NT; ++t) acc[t] = (v8f){};

  for (int k0 = 0; k0 < K; k0 += 32) {
    v16bf a;
    if constexpr (ACONTIG) {
      const float* base = Ar + k0 + asel;
      __builtin_prefetch(base + 32, 0, 1);
      float4 q0 = *(const float4*)(base);
      float4 q1 = *(const float4*)(base + 4);
      float4 q2 = *(const float4*)(base + 16);
      float4 q3 = *(const float4*)(base + 20);
      a[0]=(__bf16)q0.x; a[1]=(__bf16)q0.y; a[2]=(__bf16)q0.z; a[3]=(__bf16)q0.w;
      a[4]=(__bf16)q1.x; a[5]=(__bf16)q1.y; a[6]=(__bf16)q1.z; a[7]=(__bf16)q1.w;
      a[8]=(__bf16)q2.x; a[9]=(__bf16)q2.y; a[10]=(__bf16)q2.z; a[11]=(__bf16)q2.w;
      a[12]=(__bf16)q3.x; a[13]=(__bf16)q3.y; a[14]=(__bf16)q3.z; a[15]=(__bf16)q3.w;
    } else {
#pragma unroll
      for (int i = 0; i < 16; ++i) {
        int kk = k0 + (i < 8 ? i : i + 8) + asel;
        a[i] = (__bf16)Ar[kk * a_cs];
      }
    }
#pragma unroll
    for (int t = 0; t < NT; ++t) {
      const float* Bc = B + (size_t)(col0 + t * 16 + lid) * b_cs;
      v16bf b;
#pragma unroll
      for (int i = 0; i < 16; ++i) {
        int kk = k0 + (i < 8 ? i : i + 8) + bsel;
        b[i] = (__bf16)Bc[kk * b_rs];
      }
      acc[t] = __builtin_amdgcn_wmma_f32_16x16x32_bf16(false, a, false, b,
                                                       (short)0, acc[t], false, false);
    }
  }
  // D layout: elem r -> (M = r (+8 for hi lanes), N = lane%16)
#pragma unroll
  for (int t = 0; t < NT; ++t) {
    float* Cr = C + (size_t)(col0 + t * 16 + lid) * c_cs
                  + (size_t)(row0 + (hi ? 8 : 0)) * c_rs;
#pragma unroll
    for (int r = 0; r < 8; ++r) {
      float v = acc[t][r] * scale;
      if (act) v = siluf(v);
      Cr[r * c_rs] = v;
    }
  }
}

// ---------------- edge message scatter (outer product + atomics) ------------
__global__ void scatter_k(const float* __restrict__ x,       // [N,64]
                          const float* __restrict__ eattr,   // [E,16]
                          const float* __restrict__ mattr,   // [N,16]
                          const int* __restrict__ eidx,      // [2,E]
                          const float* __restrict__ tpw,     // [E,256]
                          const float* __restrict__ tpwm,    // [E,256]
                          float* __restrict__ msg,           // [N,64,16]
                          float* __restrict__ mmsg, int E) {
  int i = blockIdx.x * blockDim.x + threadIdx.x;
  if (i >= E * 64) return;
  int e = i >> 6, c = i & 63;
  int s = eidx[e], r = eidx[E + e];
  float xj = x[(long)s * 64 + c];
  float w[4], wm[4];
#pragma unroll
  for (int l = 0; l < 4; ++l) {
    w[l]  = tpw[(long)e * 256 + l * 64 + c] * xj;
    wm[l] = tpwm[(long)e * 256 + l * 64 + c] * xj;
  }
  const float* ea = eattr + (long)e * 16;
  const float* ma = mattr + (long)s * 16;
  float* mo  = msg  + (long)r * 1024 + c * 16;
  float* mmo = mmsg + (long)r * 1024 + c * 16;
#pragma unroll
  for (int m = 0; m < 16; ++m) {
    int l = (m == 0) ? 0 : (m < 4) ? 1 : (m < 9) ? 2 : 3;
    atomicAdd(&mo[m],  ea[m] * w[l]);
    atomicAdd(&mmo[m], ma[m] * wm[l]);
  }
}

// W_skip [L][C][A][C] -> Wt [L][32][C][C], v rows 16..31 zeroed
__global__ void transpose_skip_k(const float* __restrict__ W,
                                 float* __restrict__ Wt, int total) {
  int i = blockIdx.x * blockDim.x + threadIdx.x;
  if (i >= total) return;
  int w = i & 63, u = (i >> 6) & 63, v = (i >> 12) & 31, l = i >> 17;
  Wt[i] = (v < 16) ? W[(((long)l * 64 + u) * 16 + v) * 64 + w] : 0.f;
}

// ------------------------- skip tensor product ------------------------------
// out[n,w,m] = (1/32) * sum_u xn[n,u,m] * Weff[n,u,w]
//   where xn = x/(density+1) (mode 0) or x/16 (mode 1)
// Weff[n,u,w] = sum_v attrs[n,v] * Wt[l][v][u][w]  -> WMMA: attrs16 @ Wt[l]
// Block = 16 nodes, 4 waves; wave j owns w-strip [16j,16j+16).
// Each lane register-accumulates acc[8 nodes][d_l] over u -> no LDS RMW, and
// since l-blocks cover disjoint m-ranges the result stores straight to global.
__global__ void skip_tp_wmma(const float* __restrict__ x,       // [N,64,16]
                             const float* __restrict__ attrsp,  // [N,32] padded
                             const float* __restrict__ Wt,      // [L][32][64][64]
                             const float* __restrict__ density, // [N]
                             int mode,
                             float* __restrict__ out, int N) {
  extern __shared__ float smem[];
  float* xs = smem;             // 16*64*16 floats
  int n0   = blockIdx.x * 16;
  int tid  = threadIdx.x;       // 0..127
  int wave = tid >> 5;
  int lane = tid & 31;
  bool hi  = lane >= 16;
  int lid  = lane & 15;
  int bsel = hi ? 8 : 0;
  int nodeBase = hi ? 8 : 0;

  // stage (and normalize) the 16-node message block into LDS
  for (int i = tid; i < 16384; i += 128) {
    float v = x[(long)n0 * 1024 + i];
    if (mode == 0) v = v / (density[n0 + (i >> 10)] + 1.f);
    else           v = v * 0.0625f;
    xs[i] = v;
  }
  __syncthreads();

  // A operand: padded attrs for 16 nodes (K = 32, rows 16..31 are zero)
  v16bf a;
  {
    const float* ar = attrsp + (size_t)(n0 + lid) * 32;
#pragma unroll
    for (int i = 0; i < 16; ++i) {
      int kk = (i < 8 ? i : i + 8) + (hi ? 8 : 0);
      a[i] = (__bf16)ar[kk];
    }
  }

  int w0 = wave * 16;
  int w  = w0 + lid;

#pragma unroll
  for (int l = 0; l < 4; ++l) {
    const int mlo = (l == 0) ? 0 : (l == 1) ? 1 : (l == 2) ? 4 : 9;
    const int dl  = (l == 0) ? 1 : (l == 1) ? 3 : (l == 2) ? 5 : 7;
    const float* Wl = Wt + (size_t)l * 131072 + w0 + lid;

    float acc[8][7];
#pragma unroll
    for (int r = 0; r < 8; ++r)
#pragma unroll
      for (int m = 0; m < 7; ++m) acc[r][m] = 0.f;

#pragma unroll 2
    for (int u = 0; u < 64; ++u) {
      v16bf b;
#pragma unroll
      for (int i = 0; i < 16; ++i)
        b[i] = (__bf16)Wl[((i < 8 ? i : i + 8) + bsel) * 4096 + u * 64];
      v8f weff = {};
      weff = __builtin_amdgcn_wmma_f32_16x16x32_bf16(false, a, false, b,
                                                     (short)0, weff, false, false);
#pragma unroll
      for (int r = 0; r < 8; ++r) {
        float wv = weff[r];
        const float* xrow = xs + (nodeBase + r) * 1024 + u * 16 + mlo;
#pragma unroll
        for (int m = 0; m < dl; ++m) acc[r][m] += xrow[m] * wv;
      }
    }
    // disjoint (node, w, m-range) per lane/l -> direct global stores
#pragma unroll
    for (int r = 0; r < 8; ++r) {
      float* orow = out + (long)(n0 + nodeBase + r) * 1024 + w * 16 + mlo;
#pragma unroll
      for (int m = 0; m < dl; ++m) orow[m] = acc[r][m] * 0.03125f;
    }
  }
}

// ----------------------------- launcher -------------------------------------

extern "C" void kernel_launch(void* const* d_in, const int* in_sizes, int n_in,
                              void* d_out, int out_size, void* d_ws, size_t ws_size,
                              hipStream_t stream) {
  const float* node_attrs  = (const float*)d_in[0];
  const float* node_feats  = (const float*)d_in[1];
  const float* edge_attrs  = (const float*)d_in[2];
  const float* edge_feats  = (const float*)d_in[3];
  const int*   edge_index  = (const int*)d_in[4];
  const float* magmom_inv  = (const float*)d_in[5];
  const float* magmom_attr = (const float*)d_in[6];
  const float* W_up    = (const float*)d_in[7];
  const float* W_r0    = (const float*)d_in[8];
  const float* W_r1    = (const float*)d_in[9];
  const float* W_r2    = (const float*)d_in[10];
  const float* W_r3    = (const float*)d_in[11];
  const float* W_trans = (const float*)d_in[12];
  const float* W_dens  = (const float*)d_in[13];
  const float* W_lin   = (const float*)d_in[14];
  const float* W_mlin  = (const float*)d_in[15];
  const float* W_skip  = (const float*)d_in[16];
  const float* W_mskip = (const float*)d_in[17];
  float* out = (float*)d_out;

  const int N = in_sizes[1] / 64;   // 8192
  const int E = in_sizes[3] / 8;    // 65536

  // workspace layout (floats)
  float* ws      = (float*)d_ws;    long o = 0;
  float* x_ws    = ws + o;          o += (long)N * 64;
  float* h0      = ws + o;          o += (long)E * 32;   // K-padded
  float* h_a     = ws + o;          o += (long)E * 64;
  float* h_b     = ws + o;          o += (long)E * 64;
  float* tp_w    = ws + o;          o += (long)E * 256;  // later msg2
  float* tp_wm   = ws + o;          o += (long)E * 256;  // later msg2m
  float* density = ws + o;          o += N;
  float* msg_acc = ws + o;          o += (long)N * 1024;
  float* mm_acc  = ws + o;          o += (long)N * 1024;
  float* Wt_s    = ws + o;          o += 524288;         // [4][32][64][64]
  float* Wt_m    = ws + o;          o += 524288;
  float* W_r0p   = ws + o;          o += 32 * 64;
  float* attrs_p = ws + o;          o += (long)N * 32;

  // 1) zero accumulators (density, msg_acc, mm_acc are contiguous)
  {
    long n = (long)N + 2L * N * 1024;
    zero_k<<<dim3((n + 255) / 256), 256, 0, stream>>>(density, n);
  }

  // 2) pad small operands
  pad_rows_k<<<dim3((32 * 64 + 255) / 256), 256, 0, stream>>>(W_r0, W_r0p, 16, 32, 64);
  pad_cols_k<<<dim3((N * 32 + 255) / 256), 256, 0, stream>>>(node_attrs, attrs_p, N, 16, 32);

  // 3) linear_up: x = node_feats @ W_up / 8   (M=N, N=64 -> one 4-tile strip)
  gemm_bf16_wmma<true, 4><<<dim3(N / 64, 1), 128, 0, stream>>>(
      node_feats, 64, 1, W_up, 64, 1, x_ws, 64, 1, 64, 0.125f, 0);

  // 4) radial MLP input (K-padded to 32)
  build_h0_k<<<dim3((E * 32 + 255) / 256), 256, 0, stream>>>(
      edge_feats, magmom_inv, edge_index, h0, E);

  // 5) radial MLP + tp weights
  gemm_bf16_wmma<true, 4><<<dim3(E / 64, 1), 128, 0, stream>>>(
      h0, 32, 1, W_r0p, 64, 1, h_a, 64, 1, 32, 0.25f, 1);
  gemm_bf16_wmma<true, 4><<<dim3(E / 64, 1), 128, 0, stream>>>(
      h_a, 64, 1, W_r1, 64, 1, h_b, 64, 1, 64, 0.125f, 1);
  gemm_bf16_wmma<true, 4><<<dim3(E / 64, 1), 128, 0, stream>>>(
      h_b, 64, 1, W_r2, 64, 1, h_a, 64, 1, 64, 0.125f, 1);
  gemm_bf16_wmma<true, 4><<<dim3(E / 64, 4), 128, 0, stream>>>(
      h_a, 64, 1, W_r3, 256, 1, tp_w, 256, 1, 64, 0.125f, 0);
  gemm_bf16_wmma<true, 4><<<dim3(E / 64, 4), 128, 0, stream>>>(
      tp_w, 256, 1, W_trans, 256, 1, tp_wm, 256, 1, 256, 0.0625f, 0);

  // 6) edge density segment-sum
  density_k<<<dim3((E + 255) / 256), 256, 0, stream>>>(
      edge_feats, W_dens, edge_index, density, E);

  // 7) weighted TP outer products + scatter-add over receivers
  scatter_k<<<dim3((E * 64 + 255) / 256), 256, 0, stream>>>(
      x_ws, edge_attrs, magmom_attr, edge_index, tp_w, tp_wm,
      msg_acc, mm_acc, E);

  // 8) per-l equivariant linears: one GEMM per (m-index); reuse tp_w buffers
  float* msg2  = tp_w;
  float* msg2m = tp_wm;
  for (int m = 0; m < 16; ++m) {
    int l = (m == 0) ? 0 : (m < 4) ? 1 : (m < 9) ? 2 : 3;
    gemm_bf16_wmma<false, 4><<<dim3(N / 64, 1), 128, 0, stream>>>(
        msg_acc + m, 1024, 16, W_lin + (long)l * 4096, 64, 1,
        msg2 + m, 1024, 16, 64, 0.125f, 0);
    gemm_bf16_wmma<false, 4><<<dim3(N / 64, 1), 128, 0, stream>>>(
        mm_acc + m, 1024, 16, W_mlin + (long)l * 4096, 64, 1,
        msg2m + m, 1024, 16, 64, 0.125f, 0);
  }

  // 9) transpose + pad skip weights for WMMA-friendly layout
  transpose_skip_k<<<dim3((524288 + 255) / 256), 256, 0, stream>>>(W_skip, Wt_s, 524288);
  transpose_skip_k<<<dim3((524288 + 255) / 256), 256, 0, stream>>>(W_mskip, Wt_m, 524288);

  // 10) skip tensor products (WMMA Weff generation + fused normalize/apply)
  size_t shmem = 16384u * sizeof(float);
  skip_tp_wmma<<<dim3(N / 16), 128, shmem, stream>>>(
      msg2, attrs_p, Wt_s, density, 0, out, N);
  skip_tp_wmma<<<dim3(N / 16), 128, shmem, stream>>>(
      msg2m, attrs_p, Wt_m, density, 1, out + (long)N * 1024, N);
}
